// SiglipSelfAttention_29875792511814
// MI455X (gfx1250) — matrix operand order repaired and checked by reading, not compiled, not run on hardware
//
#include <hip/hip_runtime.h>

#define BB   8
#define SS   1024
#define DD   768
#define HH   12
#define HDIM 64

typedef __attribute__((ext_vector_type(16))) __bf16 v16bf;
typedef __attribute__((ext_vector_type(8)))  float  v8f;

union V16 { v16bf v; uint4 q[2]; };

__device__ __forceinline__ unsigned short f2bf(float f) {
  unsigned int u = __float_as_uint(f);
  u += 0x7fffu + ((u >> 16) & 1u);           // round-to-nearest-even
  return (unsigned short)(u >> 16);
}

// A operand: 16x32 bf16 tile. Per-lane: row = lane&15; elements 0..7 at
// K = k0 + (lane>>4)*8, elements 8..15 at K = k0 + 16 + (lane>>4)*8.
__device__ __forceinline__ v16bf load_A(const unsigned short* src, int ld,
                                        int row0, int k0) {
  int lane = threadIdx.x & 31;
  const unsigned short* p = src + (size_t)(row0 + (lane & 15)) * ld + k0 + (lane >> 4) * 8;
  V16 t;
  t.q[0] = *(const uint4*)(p);
  t.q[1] = *(const uint4*)(p + 16);
  return t.v;
}

// B operand: 32x16 bf16 tile from column-major storage colmaj[n*ld + k].
// Per-lane: column n = n0 + (lane&15); elements 0..7 at K = k0 + (lane>>4)*8,
// elements 8..15 at K = k0 + 16 + (lane>>4)*8.
__device__ __forceinline__ v16bf load_B(const unsigned short* colmaj, int ld,
                                        int k0, int n0) {
  int lane = threadIdx.x & 31;
  const unsigned short* p = colmaj + (size_t)(n0 + (lane & 15)) * ld + k0 + (lane >> 4) * 8;
  V16 t;
  t.q[0] = *(const uint4*)(p);
  t.q[1] = *(const uint4*)(p + 16);
  return t.v;
}

__device__ __forceinline__ v8f wmma_bf16(v16bf a, v16bf b, v8f c) {
  return __builtin_amdgcn_wmma_f32_16x16x32_bf16(false, a, false, b, (short)0, c,
                                                 false, false);
}

// Deterministic per-element dropout decision (p_keep = 0.5), key folded in.
__device__ __forceinline__ bool keep_elem(unsigned long long idx) {
  unsigned int x = (unsigned int)(idx ^ (idx >> 32)) ^ 42u;
  x *= 0x9E3779B9u; x ^= x >> 16; x *= 0x85EBCA6Bu; x ^= x >> 13;
  x *= 0xC2B2AE35u; x ^= x >> 16;
  return (x & 1u) != 0u;
}

__global__ void cvt_bf16_kernel(const float* __restrict__ src,
                                unsigned short* __restrict__ dst, int n) {
  int i = blockIdx.x * blockDim.x + threadIdx.x;
  if (i < n) dst[i] = f2bf(src[i]);
}

// ------- fused QKV projection: 16 rows x 64 cols per wave (4 accumulators) ---
// A-operand (x tile) is loaded once per k-step and reused across 4 B tiles:
// 4x less activation traffic and 4 independent WMMA accumulation chains.
__global__ __launch_bounds__(32) void qkv_kernel(
    const unsigned short* __restrict__ xb,
    const unsigned short* __restrict__ wqb, const unsigned short* __restrict__ wkb,
    const unsigned short* __restrict__ wvb,
    const float* __restrict__ bq, const float* __restrict__ bk,
    const float* __restrict__ bv,
    unsigned short* __restrict__ Qb, unsigned short* __restrict__ Kb,
    unsigned short* __restrict__ Vtb) {
  const int rowTile = blockIdx.x;           // B*S/16 tiles of rows
  const int colTile = blockIdx.y;           // D/64 tiles of output features
  const int which   = blockIdx.z;           // 0=Q 1=K 2=V
  const unsigned short* W = (which == 0) ? wqb : (which == 1) ? wkb : wvb;
  const float* bias       = (which == 0) ? bq  : (which == 1) ? bk  : bv;

  const int row0 = rowTile * 16, e0 = colTile * 64;
  v8f acc0 = {}, acc1 = {}, acc2 = {}, acc3 = {};
#pragma unroll 2
  for (int k = 0; k < DD; k += 32) {
    const v16bf a = load_A(xb, DD, row0, k);          // x rows, contiguous d
    acc0 = wmma_bf16(a, load_B(W, DD, k, e0),      acc0);
    acc1 = wmma_bf16(a, load_B(W, DD, k, e0 + 16), acc1);
    acc2 = wmma_bf16(a, load_B(W, DD, k, e0 + 32), acc2);
    acc3 = wmma_bf16(a, load_B(W, DD, k, e0 + 48), acc3);
  }

  const int lane = threadIdx.x & 31, n = lane & 15, half = lane >> 4;
  v8f accs[4] = {acc0, acc1, acc2, acc3};
#pragma unroll
  for (int g = 0; g < 4; ++g) {
    const int e = e0 + g * 16 + n, h = e / HDIM, hd = e % HDIM;
    const float bval = bias[e];
#pragma unroll
    for (int r = 0; r < 8; ++r) {
      const int m = r + 8 * half;
      const int sflat = row0 + m;
      const int b_ = sflat / SS, s = sflat % SS;
      const unsigned short val = f2bf(accs[g][r] + bval);
      if (which == 2)
        Vtb[(((size_t)b_ * HH + h) * HDIM + hd) * SS + s] = val;  // V transposed
      else {
        unsigned short* dst = (which == 0) ? Qb : Kb;
        dst[(((size_t)b_ * HH + h) * SS + s) * HDIM + hd] = val;
      }
    }
  }
}

// ------------- flash attention: one (b,h) x 16-query tile per wave ----------
__global__ __launch_bounds__(32) void attn_kernel(
    const unsigned short* __restrict__ Qb, const unsigned short* __restrict__ Kb,
    const unsigned short* __restrict__ Vtb, unsigned short* __restrict__ Ob) {
  __shared__ unsigned short ldsP[16 * 32];  // P tile staging (D-layout -> A-layout)

  const int bh = blockIdx.x, qTile = blockIdx.y;
  const int b = bh / HH, h = bh % HH;
  const unsigned short* Qh = Qb  + (((size_t)b * HH + h) * SS) * HDIM;
  const unsigned short* Kh = Kb  + (((size_t)b * HH + h) * SS) * HDIM;
  const unsigned short* Vh = Vtb + (((size_t)b * HH + h) * HDIM) * SS;

  const int lane = threadIdx.x & 31, n = lane & 15, half = lane >> 4;
  const int q0 = qTile * 16;

  const v16bf qa0 = load_A(Qh, HDIM, q0, 0);    // hd 0..31
  const v16bf qa1 = load_A(Qh, HDIM, q0, 32);   // hd 32..63

  v8f o0 = {}, o1 = {}, o2 = {}, o3 = {};
  float rm[8], rl[8];
#pragma unroll
  for (int r = 0; r < 8; ++r) { rm[r] = -3.0e38f; rl[r] = 0.0f; }

  const float sc = 0.125f;                  // 1/sqrt(64)
  for (int kt = 0; kt < SS; kt += 32) {
    if (kt + 32 < SS) {
      __builtin_prefetch(Kh + (size_t)(kt + 32) * HDIM, 0, 1);
      __builtin_prefetch(Vh + (size_t)(kt + 32), 0, 1);
    }

    // scores S[16q x 32k] as two 16x16 C/D tiles
    v8f s0 = {}, s1 = {};
    s0 = wmma_bf16(qa0, load_B(Kh, HDIM, 0,  kt),      s0);
    s0 = wmma_bf16(qa1, load_B(Kh, HDIM, 32, kt),      s0);
    s1 = wmma_bf16(qa0, load_B(Kh, HDIM, 0,  kt + 16), s1);
    s1 = wmma_bf16(qa1, load_B(Kh, HDIM, 32, kt + 16), s1);

    float sv[16];
#pragma unroll
    for (int r = 0; r < 8; ++r) {           // scale + dropout (before softmax)
      const int m = r + 8 * half;
      const unsigned long long i0 =
          ((((unsigned long long)bh) * SS + (q0 + m)) * SS) + (kt + n);
      float x0 = s0[r] * sc, x1 = s1[r] * sc;
      sv[r]     = keep_elem(i0)      ? x0 * 2.0f : 0.0f;
      sv[r + 8] = keep_elem(i0 + 16) ? x1 * 2.0f : 0.0f;
    }

#pragma unroll
    for (int r = 0; r < 8; ++r) {           // online softmax per row
      float mx = fmaxf(sv[r], sv[r + 8]);
      mx = fmaxf(mx, __shfl_xor(mx, 1, 32));
      mx = fmaxf(mx, __shfl_xor(mx, 2, 32));
      mx = fmaxf(mx, __shfl_xor(mx, 4, 32));
      mx = fmaxf(mx, __shfl_xor(mx, 8, 32));
      const float mnew  = fmaxf(rm[r], mx);
      const float alpha = __expf(rm[r] - mnew);
      rm[r] = mnew;
      const float p0 = __expf(sv[r]     - mnew);
      const float p1 = __expf(sv[r + 8] - mnew);
      float rs = p0 + p1;
      rs += __shfl_xor(rs, 1, 32);
      rs += __shfl_xor(rs, 2, 32);
      rs += __shfl_xor(rs, 4, 32);
      rs += __shfl_xor(rs, 8, 32);
      rl[r] = rl[r] * alpha + rs;
      o0[r] *= alpha; o1[r] *= alpha; o2[r] *= alpha; o3[r] *= alpha;
      sv[r] = p0; sv[r + 8] = p1;
    }

    // D-layout -> A-layout for P via LDS (wave-private; DS ops are in-order,
    // s_wait_dscnt 0 guards the VGPR reload)
#pragma unroll
    for (int r = 0; r < 8; ++r) {
      const int m = r + 8 * half;
      ldsP[m * 32 + n]      = f2bf(sv[r]);
      ldsP[m * 32 + 16 + n] = f2bf(sv[r + 8]);
    }
    asm volatile("s_wait_dscnt 0" ::: "memory");
    const v16bf pa = load_A(ldsP, 32, 0, 0);

    // O += P @ V  (V^T storage makes B loads contiguous)
    o0 = wmma_bf16(pa, load_B(Vh, SS, kt, 0),  o0);
    o1 = wmma_bf16(pa, load_B(Vh, SS, kt, 16), o1);
    o2 = wmma_bf16(pa, load_B(Vh, SS, kt, 32), o2);
    o3 = wmma_bf16(pa, load_B(Vh, SS, kt, 48), o3);
  }

#pragma unroll
  for (int r = 0; r < 8; ++r) {
    const float inv = 1.0f / rl[r];
    const int m = r + 8 * half;
    const size_t base = (((size_t)b) * SS + (q0 + m)) * DD + h * HDIM + n;
    Ob[base + 0]  = f2bf(o0[r] * inv);
    Ob[base + 16] = f2bf(o1[r] * inv);
    Ob[base + 32] = f2bf(o2[r] * inv);
    Ob[base + 48] = f2bf(o3[r] * inv);
  }
}

// ---------------- output projection: 16x64 tile per wave ----------------
__global__ __launch_bounds__(32) void oproj_kernel(
    const unsigned short* __restrict__ Ob, const unsigned short* __restrict__ wob,
    const float* __restrict__ bo, float* __restrict__ out) {
  const int row0 = blockIdx.x * 16, e0 = blockIdx.y * 64;
  v8f acc0 = {}, acc1 = {}, acc2 = {}, acc3 = {};
#pragma unroll 2
  for (int k = 0; k < DD; k += 32) {
    const v16bf a = load_A(Ob, DD, row0, k);
    acc0 = wmma_bf16(a, load_B(wob, DD, k, e0),      acc0);
    acc1 = wmma_bf16(a, load_B(wob, DD, k, e0 + 16), acc1);
    acc2 = wmma_bf16(a, load_B(wob, DD, k, e0 + 32), acc2);
    acc3 = wmma_bf16(a, load_B(wob, DD, k, e0 + 48), acc3);
  }
  const int lane = threadIdx.x & 31, n = lane & 15, half = lane >> 4;
  v8f accs[4] = {acc0, acc1, acc2, acc3};
#pragma unroll
  for (int g = 0; g < 4; ++g) {
    const float bval = bo[e0 + g * 16 + n];
#pragma unroll
    for (int r = 0; r < 8; ++r) {
      const int m = r + 8 * half;
      out[((size_t)(row0 + m)) * DD + e0 + g * 16 + n] = accs[g][r] + bval;
    }
  }
}

extern "C" void kernel_launch(void* const* d_in, const int* in_sizes, int n_in,
                              void* d_out, int out_size, void* d_ws, size_t ws_size,
                              hipStream_t stream) {
  const float* x  = (const float*)d_in[0];
  const float* wq = (const float*)d_in[1];
  const float* bq = (const float*)d_in[2];
  const float* wk = (const float*)d_in[3];
  const float* bk = (const float*)d_in[4];
  const float* wv = (const float*)d_in[5];
  const float* bv = (const float*)d_in[6];
  const float* wo = (const float*)d_in[7];
  const float* bo = (const float*)d_in[8];
  float* out = (float*)d_out;

  const size_t nx = (size_t)BB * SS * DD;   // 6,291,456
  const size_t nw = (size_t)DD * DD;        //   589,824
  unsigned short* xb  = (unsigned short*)d_ws;
  unsigned short* wqb = xb  + nx;
  unsigned short* wkb = wqb + nw;
  unsigned short* wvb = wkb + nw;
  unsigned short* wob = wvb + nw;
  unsigned short* Qb  = wob + nw;
  unsigned short* Kb  = Qb  + nx;
  unsigned short* Vtb = Kb  + nx;
  unsigned short* Ob  = Vtb + nx;           // total ~67.7 MB of d_ws

  const int cb = 256;
  cvt_bf16_kernel<<<(int)((nx + cb - 1) / cb), cb, 0, stream>>>(x,  xb,  (int)nx);
  cvt_bf16_kernel<<<(int)((nw + cb - 1) / cb), cb, 0, stream>>>(wq, wqb, (int)nw);
  cvt_bf16_kernel<<<(int)((nw + cb - 1) / cb), cb, 0, stream>>>(wk, wkb, (int)nw);
  cvt_bf16_kernel<<<(int)((nw + cb - 1) / cb), cb, 0, stream>>>(wv, wvb, (int)nw);
  cvt_bf16_kernel<<<(int)((nw + cb - 1) / cb), cb, 0, stream>>>(wo, wob, (int)nw);

  dim3 gq(BB * SS / 16, DD / 64, 3);        // 512 x 12 x 3
  qkv_kernel<<<gq, 32, 0, stream>>>(xb, wqb, wkb, wvb, bq, bk, bv, Qb, Kb, Vtb);

  dim3 ga(BB * HH, SS / 16);                // 96 x 64
  attn_kernel<<<ga, 32, 0, stream>>>(Qb, Kb, Vtb, Ob);

  dim3 go(BB * SS / 16, DD / 64);           // 512 x 12
  oproj_kernel<<<go, 32, 0, stream>>>(Ob, wob, bo, out);
}
